// QuantumFeatureExtractor_11630771437663
// MI455X (gfx1250) — matrix operand order repaired and checked by reading, not compiled, not run on hardware
//
#include <hip/hip_runtime.h>
#include <hip/hip_bf16.h>

// ---------------------------------------------------------------------------
// Fused QuantumFeatureExtractor for MI455X (gfx1250, wave32, WMMA).
//   out = relu(relu(x@W1+b1)@W2+b2) -> quantum_state -> relu(@W3+b3) @ W4 + b4
// One block = 128 rows, 8 waves; each wave owns a 16-row stripe.
// GEMMs: v_wmma_f32_16x16x32_bf16, A-fragments hoisted to registers,
// N-tiles processed in pairs (dual accumulation chains for ILP).
// ---------------------------------------------------------------------------

typedef __attribute__((ext_vector_type(16))) __bf16 v16bf;
typedef __attribute__((ext_vector_type(8)))  float  v8f;

#define BT      128            // rows per block
#define APITCH  264            // bf16 pitch for activation rows (bank-conflict pad)
#define WPITCH  264            // bf16 pitch for transposed-weight rows
#define HDIM    256
#define FDIM    512
#define NQ      8

#define ACT_ELEMS (BT * APITCH)        // 33792 bf16
#define WB_ELEMS  (HDIM * WPITCH)      // 67584 bf16
// 2*33792*2 + 67584*2 + 256*4 = 271360 bytes  (<= 320KB WGP LDS)
#define SMEM_BYTES ((2 * ACT_ELEMS + WB_ELEMS) * 2 + 256 * 4)

union FragU { uint4 q[2]; v16bf v; };
union PackU { __bf16 h[16]; uint4 q[2]; };

// A fragment (16x32 bf16): lane L holds row M=L%16; lanes 0-15: K {0..7,16..23},
// lanes 16-31: K {8..15,24..31}  -> two 16B chunks per lane.
__device__ __forceinline__ v16bf load_a(const __bf16* aBase, int lane, int kt) {
  const __bf16* p = aBase + (lane & 15) * APITCH + kt * 32 + ((lane >> 4) << 3);
  FragU f;
  f.q[0] = *(const uint4*)(p);
  f.q[1] = *(const uint4*)(p + 16);
  return f.v;
}

// B fragment (32x16 bf16), weights staged transposed (row = N, contiguous K):
// lane L holds col N=L%16; lanes 0-15: K {0..7,16..23}, lanes 16-31: K {8..15,24..31}.
__device__ __forceinline__ v16bf load_b(const __bf16* wTile, int lane, int kt) {
  const __bf16* p = wTile + (lane & 15) * WPITCH + kt * 32 + ((lane >> 4) << 3);
  FragU f;
  f.q[0] = *(const uint4*)(p);
  f.q[1] = *(const uint4*)(p + 16);
  return f.v;
}

// Stage a 256-column slab of W (256 K-rows, row-major, row pitch ldw) into LDS
// as W^T bf16: wBuf[n*WPITCH + k].  float4 global reads (global_load_b128).
__device__ __forceinline__ void stage_w(const float* __restrict__ W, int ldw,
                                        int n0, __bf16* wBuf) {
  for (int idx = threadIdx.x; idx < (HDIM * HDIM) / 4; idx += 256) {
    int k  = idx >> 6;              // 64 float4 per 256-wide row
    int n4 = (idx & 63) << 2;
    float4 w = *(const float4*)(W + k * ldw + n0 + n4);
    wBuf[(n4 + 0) * WPITCH + k] = (__bf16)w.x;
    wBuf[(n4 + 1) * WPITCH + k] = (__bf16)w.y;
    wBuf[(n4 + 2) * WPITCH + k] = (__bf16)w.z;
    wBuf[(n4 + 3) * WPITCH + k] = (__bf16)w.w;
  }
}

// One wave: 16-row x 256-col GEMM tile. A-fragments preloaded to registers,
// N-tiles in pairs (two independent WMMA chains). bias+ReLU, bf16 -> LDS.
template <int KT>
__device__ __forceinline__ void gemm_lds(const __bf16* aBase, const __bf16* wBuf,
                                         const float* bias, __bf16* oBase, int lane) {
  v16bf aF[KT];
#pragma unroll
  for (int kt = 0; kt < KT; ++kt) aF[kt] = load_a(aBase, lane, kt);

  const int nn = lane & 15;
  const int ro = (lane >> 4) << 3;   // C layout: VGPR r -> M = r (+8 for hi lanes)
  for (int nt = 0; nt < 16; nt += 2) {
    v8f acc0 = {0.f, 0.f, 0.f, 0.f, 0.f, 0.f, 0.f, 0.f};
    v8f acc1 = {0.f, 0.f, 0.f, 0.f, 0.f, 0.f, 0.f, 0.f};
    const __bf16* wT0 = wBuf + (nt    ) * 16 * WPITCH;
    const __bf16* wT1 = wBuf + (nt + 1) * 16 * WPITCH;
#pragma unroll
    for (int kt = 0; kt < KT; ++kt) {
      v16bf b0 = load_b(wT0, lane, kt);
      v16bf b1 = load_b(wT1, lane, kt);
      acc0 = __builtin_amdgcn_wmma_f32_16x16x32_bf16(false, aF[kt], false, b0,
                                                     (short)0, acc0, false, false);
      acc1 = __builtin_amdgcn_wmma_f32_16x16x32_bf16(false, aF[kt], false, b1,
                                                     (short)0, acc1, false, false);
    }
    const float bv0 = bias[nt * 16 + nn];
    const float bv1 = bias[nt * 16 + 16 + nn];
#pragma unroll
    for (int r = 0; r < 8; ++r) {
      float v0 = fmaxf(acc0[r] + bv0, 0.0f);
      float v1 = fmaxf(acc1[r] + bv1, 0.0f);
      oBase[(ro + r) * APITCH + nt * 16 + nn]      = (__bf16)v0;
      oBase[(ro + r) * APITCH + nt * 16 + 16 + nn] = (__bf16)v1;
    }
  }
}

// Final layer tile: bias (no ReLU), fp32 stores straight to HBM.
__device__ __forceinline__ void gemm_out(const __bf16* aBase, const __bf16* wBuf,
                                         const float* bias, float* oBase, int lane) {
  v16bf aF[8];
#pragma unroll
  for (int kt = 0; kt < 8; ++kt) aF[kt] = load_a(aBase, lane, kt);

  const int nn = lane & 15;
  const int ro = (lane >> 4) << 3;
  for (int nt = 0; nt < 16; nt += 2) {
    v8f acc0 = {0.f, 0.f, 0.f, 0.f, 0.f, 0.f, 0.f, 0.f};
    v8f acc1 = {0.f, 0.f, 0.f, 0.f, 0.f, 0.f, 0.f, 0.f};
    const __bf16* wT0 = wBuf + (nt    ) * 16 * WPITCH;
    const __bf16* wT1 = wBuf + (nt + 1) * 16 * WPITCH;
#pragma unroll
    for (int kt = 0; kt < 8; ++kt) {
      v16bf b0 = load_b(wT0, lane, kt);
      v16bf b1 = load_b(wT1, lane, kt);
      acc0 = __builtin_amdgcn_wmma_f32_16x16x32_bf16(false, aF[kt], false, b0,
                                                     (short)0, acc0, false, false);
      acc1 = __builtin_amdgcn_wmma_f32_16x16x32_bf16(false, aF[kt], false, b1,
                                                     (short)0, acc1, false, false);
    }
    const float bv0 = bias[nt * 16 + nn];
    const float bv1 = bias[nt * 16 + 16 + nn];
#pragma unroll
    for (int r = 0; r < 8; ++r) {
      oBase[(ro + r) * FDIM + nt * 16 + nn]      = acc0[r] + bv0;
      oBase[(ro + r) * FDIM + nt * 16 + 16 + nn] = acc1[r] + bv1;
    }
  }
}

__global__ __launch_bounds__(256, 1)
void qfe_fused(const float* __restrict__ x,
               const float* __restrict__ W1, const float* __restrict__ b1,
               const float* __restrict__ W2, const float* __restrict__ b2,
               const float* __restrict__ W3, const float* __restrict__ b3,
               const float* __restrict__ W4, const float* __restrict__ b4,
               float* __restrict__ out) {
  extern __shared__ __align__(16) char smem[];
  __bf16* actA  = (__bf16*)smem;            // ping
  __bf16* actB  = actA + ACT_ELEMS;         // pong
  __bf16* wBuf  = actB + ACT_ELEMS;         // W^T slab (bf16)
  float*  biasS = (float*)(wBuf + WB_ELEMS);

  const int tid  = threadIdx.x;
  const int lane = tid & 31;
  const int wave = tid >> 5;
  const int row0 = blockIdx.x * BT;
  const int wrow = wave * 16 * APITCH;      // wave's 16-row stripe in act buffers

  // ---- stage x (K padded 8->32) into actB, W1^T zero-padded, b1 ----
  for (int idx = tid; idx < BT * 2; idx += 256) {       // 2 float4 per row
    int r = idx >> 1, c4 = (idx & 1) << 2;
    float4 v = *(const float4*)(x + (row0 + r) * NQ + c4);
    __bf16* dst = actB + r * APITCH + c4;
    dst[0] = (__bf16)v.x; dst[1] = (__bf16)v.y;
    dst[2] = (__bf16)v.z; dst[3] = (__bf16)v.w;
  }
  for (int idx = tid; idx < BT * 24; idx += 256) {      // zero pad k=8..31
    int r = idx / 24, k = 8 + idx % 24;
    actB[r * APITCH + k] = (__bf16)0.0f;
  }
  for (int idx = tid; idx < HDIM * 32; idx += 256) {
    int n = idx >> 5, k = idx & 31;
    wBuf[n * WPITCH + k] = (k < NQ) ? (__bf16)W1[k * HDIM + n] : (__bf16)0.0f;
  }
  if (tid < HDIM) biasS[tid] = b1[tid];
  __builtin_prefetch(W2 + tid * 64, 0, 1);   // warm W2 in L2 (global_prefetch_b8)
  __syncthreads();

  // ---- L1: h = relu(x @ W1 + b1)  (K=32, one WMMA step) ----
  gemm_lds<1>(actB + wrow, wBuf, biasS, actA + wrow, lane);
  __syncthreads();

  // ---- stage W2^T + b2 ----
  stage_w(W2, HDIM, 0, wBuf);
  if (tid < HDIM) biasS[tid] = b2[tid];
  __builtin_prefetch(W3 + tid * 64, 0, 1);
  __syncthreads();

  // ---- L2: enc = relu(h @ W2 + b2) ----
  gemm_lds<8>(actA + wrow, wBuf, biasS, actB + wrow, lane);
  __syncthreads();

  // ---- quantum feature map: state[b, j] = prod_i (bit_i(j) ? sin : cos)(enc_i*pi/2)
  //      qubit 0 is the MSB of j.  Each row split across 2 threads: the top-qubit
  //      bit is tid&1 (uniform select, keeps all tables in registers).
  {
    const int r  = tid >> 1;
    const int jb = (tid & 1) << 3;             // hi-nibble base: 0 or 8
    const __bf16* erow = actB + r * APITCH;
    float c8[NQ], s8[NQ];
#pragma unroll
    for (int i = 0; i < NQ; ++i) {
      float th = (float)erow[i] * 1.57079632679489662f;  // enc * pi/2
      __sincosf(th, &s8[i], &c8[i]);
    }
    const float top = (tid & 1) ? s8[0] : c8[0];         // qubit 0 (MSB)
    float ph[8];
#pragma unroll
    for (int m = 0; m < 8; ++m) {              // hi nibble = jb + m; qubits 1..3
      float a = top;
#pragma unroll
      for (int i = 1; i < 4; ++i) {
        const int bit = (m >> (3 - i)) & 1;
        a *= bit ? s8[i] : c8[i];
      }
      ph[m] = a;
    }
    float pl[16];
#pragma unroll
    for (int m = 0; m < 16; ++m) {             // lo nibble; qubits 4..7
      float b = 1.0f;
#pragma unroll
      for (int i = 0; i < 4; ++i) {
        const int bit = (m >> (3 - i)) & 1;
        b *= bit ? s8[4 + i] : c8[4 + i];
      }
      pl[m] = b;
    }
    __bf16* srow = actA + r * APITCH + (jb << 4);        // this thread's 128 cols
#pragma unroll
    for (int hi = 0; hi < 8; ++hi) {
      PackU pk;
#pragma unroll
      for (int lo = 0; lo < 16; ++lo)
        pk.h[lo] = (__bf16)(ph[hi] * pl[lo]);
      *(uint4*)(srow + hi * 16)     = pk.q[0];
      *(uint4*)(srow + hi * 16 + 8) = pk.q[1];
    }
  }
  __syncthreads();

  // ---- stage W3^T + b3 ----
  stage_w(W3, HDIM, 0, wBuf);
  if (tid < HDIM) biasS[tid] = b3[tid];
  __builtin_prefetch(W4 + tid * 256, 0, 1);
  __syncthreads();

  // ---- L3: f = relu(state @ W3 + b3) ----
  gemm_lds<8>(actA + wrow, wBuf, biasS, actB + wrow, lane);
  __syncthreads();

  // ---- L4: out = f @ W4 + b4, in two 256-column halves ----
  for (int half = 0; half < 2; ++half) {
    stage_w(W4, FDIM, half * HDIM, wBuf);
    if (tid < HDIM) biasS[tid] = b4[half * HDIM + tid];
    __syncthreads();
    gemm_out(actB + wrow, wBuf, biasS,
             out + (row0 + wave * 16) * FDIM + half * HDIM, lane);
    __syncthreads();
  }
}

extern "C" void kernel_launch(void* const* d_in, const int* in_sizes, int n_in,
                              void* d_out, int out_size, void* d_ws, size_t ws_size,
                              hipStream_t stream) {
  const float* x  = (const float*)d_in[0];
  const float* W1 = (const float*)d_in[1];
  const float* b1 = (const float*)d_in[2];
  const float* W2 = (const float*)d_in[3];
  const float* b2 = (const float*)d_in[4];
  const float* W3 = (const float*)d_in[5];
  const float* b3 = (const float*)d_in[6];
  const float* W4 = (const float*)d_in[7];
  const float* b4 = (const float*)d_in[8];
  float* out = (float*)d_out;

  // Allow 265KB dynamic LDS (deterministic, idempotent; not a stream op).
  (void)hipFuncSetAttribute((const void*)qfe_fused,
                            hipFuncAttributeMaxDynamicSharedMemorySize,
                            (int)SMEM_BYTES);

  const int B = in_sizes[0] / NQ;           // 131072
  dim3 grid(B / BT), block(256);
  qfe_fused<<<grid, block, SMEM_BYTES, stream>>>(x, W1, b1, W2, b2, W3, b3, W4, b4, out);
}
